// ResonantBrain_32641751449696
// MI455X (gfx1250) — matrix operand order repaired and checked by reading, not compiled, not run on hardware
//
#include <hip/hip_runtime.h>
#include <hip/hip_bf16.h>
#include <cstddef>

#define L_SEQ 2048
#define D_MOD 256
#define N_HEAD 4
#define HD 64
#define N_LAYER 6
#define M_MEM 8192
#define V_VOCAB 50257

typedef __attribute__((ext_vector_type(16))) __bf16 v16bf;
typedef __attribute__((ext_vector_type(8)))  float  v8f;

// ---------------------------------------------------------------------------
// WMMA helpers (CDNA5 gfx1250, wave32). Layouts per ISA 7.12.2:
//  A (16x32, 16-bit): lane l holds row (l&15); VGPR j holds K = base(j)+8*(l>>4)
//    base(j) = 2j (j<4) else 16+2(j-4)
//  B (32x16, 16-bit): lane l holds col (l&15); same K order as A:
//    VGPR j holds K = base(j)+8*(l>>4), +1
//  C/D (16x16 f32): lane l holds col (l&15); VGPR i holds row i+8*(l>>4)
// ---------------------------------------------------------------------------
__device__ inline v8f wmma_bf16(v16bf a, v16bf b, v8f c) {
  return __builtin_amdgcn_wmma_f32_16x16x32_bf16(false, a, false, b, (short)0, c,
                                                 false, false);
}

__device__ inline v16bf load_a_bf16(const __bf16* A, int ld, int row0, int k0) {
  int l = threadIdx.x & 31;
  int r = row0 + (l & 15);
  int g = l >> 4;
  const __bf16* p = A + (size_t)r * ld + k0 + 8 * g;
  v16bf f;
#pragma unroll
  for (int j = 0; j < 8; ++j) {
    int kb = (j < 4) ? (2 * j) : (16 + 2 * (j - 4));
    f[2 * j]     = p[kb];
    f[2 * j + 1] = p[kb + 1];
  }
  return f;
}

// A-fragment from an LDS f32 tile of shape [16][32] (row-major), cvt -> bf16
__device__ inline v16bf load_a_lds32(const float* t) {
  int l = threadIdx.x & 31;
  const float* p = t + (l & 15) * 32 + 8 * (l >> 4);
  v16bf f;
#pragma unroll
  for (int j = 0; j < 8; ++j) {
    int kb = (j < 4) ? (2 * j) : (16 + 2 * (j - 4));
    f[2 * j]     = (__bf16)p[kb];
    f[2 * j + 1] = (__bf16)p[kb + 1];
  }
  return f;
}

// B-fragment where source is row-major [N,K] and we need B[k][n] = S[n][k]
template <typename T>
__device__ inline v16bf load_bT(const T* S, int ld, int n0, int k0, int nmax) {
  int l = threadIdx.x & 31;
  int n = n0 + (l & 15);
  if (n > nmax) n = nmax;
  const T* p = S + (size_t)n * ld + k0 + 8 * (l >> 4);
  v16bf f;
#pragma unroll
  for (int j = 0; j < 8; ++j) {
    int kb = (j < 4) ? (2 * j) : (16 + 2 * (j - 4));
    f[2 * j]     = (__bf16)p[kb];
    f[2 * j + 1] = (__bf16)p[kb + 1];
  }
  return f;
}

// B-fragment where source is row-major [K,N]: B[k][n] = S[k0+k][n0+n]
template <typename T>
__device__ inline v16bf load_bKN(const T* S, int ld, int k0, int n0) {
  int l = threadIdx.x & 31;
  const T* p = S + (size_t)(k0 + 8 * (l >> 4)) * ld + n0 + (l & 15);
  v16bf f;
#pragma unroll
  for (int j = 0; j < 8; ++j) {
    int kb = (j < 4) ? (2 * j) : (16 + 2 * (j - 4));
    f[2 * j]     = (__bf16)p[(size_t)kb * ld];
    f[2 * j + 1] = (__bf16)p[(size_t)(kb + 1) * ld];
  }
  return f;
}

__device__ inline float redmax16(float v) {
#pragma unroll
  for (int m = 1; m < 16; m <<= 1) v = fmaxf(v, __shfl_xor(v, m, 32));
  return v;
}
__device__ inline float redsum16(float v) {
#pragma unroll
  for (int m = 1; m < 16; m <<= 1) v += __shfl_xor(v, m, 32);
  return v;
}

// ---------------------------------------------------------------------------
// Elementwise / small kernels
// ---------------------------------------------------------------------------
__global__ void k_gather(const int* tok, const float* emb, float* x) {
  int l = blockIdx.x, t = threadIdx.x;
  x[(size_t)l * D_MOD + t] = emb[(size_t)tok[l] * D_MOD + t];
}

__global__ void k_rmsnorm(const float* x, const float* w, __bf16* o) {
  __shared__ float red[D_MOD];
  int l = blockIdx.x, t = threadIdx.x;
  float xv = x[(size_t)l * D_MOD + t];
  red[t] = xv * xv;
  __syncthreads();
  for (int s = D_MOD / 2; s > 0; s >>= 1) {
    if (t < s) red[t] += red[t + s];
    __syncthreads();
  }
  float sc = rsqrtf(red[0] / (float)D_MOD + 1e-6f);
  o[(size_t)l * D_MOD + t] = (__bf16)(xv * sc * w[t]);
}

// qkv [L,768] (f32) -> rope'd per-head bf16 q/k and bf16 v, layout [H][L][HD]
__global__ void k_rope_split(const float* qkv, __bf16* qh, __bf16* kh, __bf16* vh) {
  int idx = blockIdx.x * blockDim.x + threadIdx.x;  // L*H*32 pairs
  if (idx >= L_SEQ * N_HEAD * (HD / 2)) return;
  int i = idx & 31;
  int h = (idx >> 5) & (N_HEAD - 1);
  int pos = idx >> 7;
  float inv = __powf(10000.f, -(float)(2 * i) / (float)HD);
  float ang = (float)pos * inv;
  float c = __cosf(ang), s = __sinf(ang);
  const float* b = qkv + (size_t)pos * (3 * D_MOD);
  int co = h * HD + 2 * i;
  float q0 = b[co], q1 = b[co + 1];
  float k0 = b[D_MOD + co], k1 = b[D_MOD + co + 1];
  float v0 = b[2 * D_MOD + co], v1 = b[2 * D_MOD + co + 1];
  size_t o = ((size_t)h * L_SEQ + pos) * HD + 2 * i;
  qh[o] = (__bf16)(q0 * c - q1 * s);
  qh[o + 1] = (__bf16)(q0 * s + q1 * c);
  kh[o] = (__bf16)(k0 * c - k1 * s);
  kh[o + 1] = (__bf16)(k0 * s + k1 * c);
  vh[o] = (__bf16)v0;
  vh[o + 1] = (__bf16)v1;
}

// ---------------------------------------------------------------------------
// Generic WMMA GEMM: out[M,N] = A[M,K](bf16) @ W[N,K](f32)^T (+bias)(GELU)(+res)
// Each wave computes a 16x32 strip (2 B-frags per A-frag -> 2x wmma density
// per A-side load).  block = 256 thr = 8 waves; grid = ((N+255)/256, M/16).
// ---------------------------------------------------------------------------
__global__ void k_gemm(const __bf16* A, const float* W, const float* bias,
                       const float* res, float* outF, __bf16* outB,
                       int M, int N, int K, int gelu) {
  int wave = threadIdx.x >> 5;
  int l = threadIdx.x & 31;
  int m0 = blockIdx.y * 16;
  int n0 = blockIdx.x * 256 + wave * 32;
  if (n0 >= N) return;
  v8f acc0 = {0, 0, 0, 0, 0, 0, 0, 0}, acc1 = acc0;
  for (int k0 = 0; k0 < K; k0 += 32) {
    if (k0 + 32 < K) {  // prefetch next K-tile (global_prefetch_b8)
      __builtin_prefetch(A + (size_t)(m0 + (l & 15)) * K + k0 + 32, 0, 3);
      __builtin_prefetch(W + (size_t)min(n0 + (l & 15), N - 1) * K + k0 + 32, 0, 3);
      __builtin_prefetch(W + (size_t)min(n0 + 16 + (l & 15), N - 1) * K + k0 + 32, 0, 3);
    }
    v16bf a = load_a_bf16(A, K, m0, k0);
    v16bf b0 = load_bT<float>(W, K, n0, k0, N - 1);
    v16bf b1 = load_bT<float>(W, K, n0 + 16, k0, N - 1);
    acc0 = wmma_bf16(a, b0, acc0);
    acc1 = wmma_bf16(a, b1, acc1);
  }
  int g = l >> 4;
#pragma unroll
  for (int half = 0; half < 2; ++half) {
    int n = n0 + 16 * half + (l & 15);
    if (n >= N) continue;
    float bv = bias ? bias[n] : 0.f;
    v8f acc = half ? acc1 : acc0;
#pragma unroll
    for (int i = 0; i < 8; ++i) {
      int row = m0 + i + 8 * g;
      float v = acc[i] + bv;
      if (gelu) v = 0.5f * v * (1.f + erff(v * 0.70710678f));
      if (res) v += res[(size_t)row * N + n];
      if (outF) outF[(size_t)row * N + n] = v;
      if (outB) outB[(size_t)row * N + n] = (__bf16)v;
    }
  }
}

// ---------------------------------------------------------------------------
// Flash attention, causal, hd=64.  block=128 (4 waves, 1 q-tile each),
// grid = (L/64, H).  P repacked via per-wave LDS tile (16x32 f32).
// ---------------------------------------------------------------------------
__global__ void k_attn(const __bf16* qh, const __bf16* kh, const __bf16* vh,
                       __bf16* out) {
  __shared__ float pbuf[4][16 * 32];
  int wave = threadIdx.x >> 5;
  int l = threadIdx.x & 31;
  int g = l >> 4;
  int col = l & 15;
  int h = blockIdx.y;
  int q0 = (blockIdx.x * 4 + wave) * 16;
  const __bf16* Q = qh + (size_t)h * L_SEQ * HD;
  const __bf16* Km = kh + (size_t)h * L_SEQ * HD;
  const __bf16* Vm = vh + (size_t)h * L_SEQ * HD;
  float* pt = pbuf[wave];

  v16bf qa0 = load_a_bf16(Q, HD, q0, 0);
  v16bf qa1 = load_a_bf16(Q, HD, q0, 32);
  v8f o0 = {0, 0, 0, 0, 0, 0, 0, 0}, o1 = o0, o2 = o0, o3 = o0;
  float mrun[8], srun[8];
#pragma unroll
  for (int i = 0; i < 8; ++i) { mrun[i] = -1e30f; srun[i] = 0.f; }

  for (int kb = 0; kb < q0 + 16; kb += 32) {
    if (kb + 32 < q0 + 16) {  // prefetch next key/value rows
      __builtin_prefetch(Km + (size_t)(kb + 32 + l) * HD, 0, 3);
      __builtin_prefetch(Vm + (size_t)(kb + 32 + l) * HD, 0, 3);
    }
    v8f s0 = {0, 0, 0, 0, 0, 0, 0, 0}, s1 = s0;
    s0 = wmma_bf16(qa0, load_bT<__bf16>(Km, HD, kb, 0, L_SEQ - 1), s0);
    s0 = wmma_bf16(qa1, load_bT<__bf16>(Km, HD, kb, 32, L_SEQ - 1), s0);
    s1 = wmma_bf16(qa0, load_bT<__bf16>(Km, HD, kb + 16, 0, L_SEQ - 1), s1);
    s1 = wmma_bf16(qa1, load_bT<__bf16>(Km, HD, kb + 16, 32, L_SEQ - 1), s1);
#pragma unroll
    for (int i = 0; i < 8; ++i) {
      int row = q0 + i + 8 * g;
      float a = s0[i] * 0.125f;
      float b = s1[i] * 0.125f;
      if (kb + col > row) a = -1e30f;
      if (kb + 16 + col > row) b = -1e30f;
      float mx = redmax16(fmaxf(a, b));
      float mnew = fmaxf(mrun[i], mx);
      float p0 = __expf(a - mnew);
      float p1 = __expf(b - mnew);
      float rs = redsum16(p0 + p1);
      float alpha = __expf(mrun[i] - mnew);
      srun[i] = srun[i] * alpha + rs;
      o0[i] *= alpha; o1[i] *= alpha; o2[i] *= alpha; o3[i] *= alpha;
      mrun[i] = mnew;
      pt[(i + 8 * g) * 32 + col] = p0;
      pt[(i + 8 * g) * 32 + col + 16] = p1;
    }
    asm volatile("s_wait_dscnt 0" ::: "memory");  // wave-local LDS RAW fence
    v16bf pa = load_a_lds32(pt);
    o0 = wmma_bf16(pa, load_bKN<__bf16>(Vm, HD, kb, 0), o0);
    o1 = wmma_bf16(pa, load_bKN<__bf16>(Vm, HD, kb, 16), o1);
    o2 = wmma_bf16(pa, load_bKN<__bf16>(Vm, HD, kb, 32), o2);
    o3 = wmma_bf16(pa, load_bKN<__bf16>(Vm, HD, kb, 48), o3);
  }
#pragma unroll
  for (int i = 0; i < 8; ++i) {
    int row = q0 + i + 8 * g;
    float inv = 1.f / srun[i];
    size_t o = (size_t)row * D_MOD + h * HD + col;
    out[o]      = (__bf16)(o0[i] * inv);
    out[o + 16] = (__bf16)(o1[i] * inv);
    out[o + 32] = (__bf16)(o2[i] * inv);
    out[o + 48] = (__bf16)(o3[i] * inv);
  }
}

// ---------------------------------------------------------------------------
// Landmark-memory prep: per row of x -> qn(bf16, L2-normalized), need/sal/gate
// ---------------------------------------------------------------------------
__global__ void k_rowops(const float* x, const float* unc_w, const float* unc_b,
                         const float* sal_w, const float* sal_b,
                         const float* wg_w, const float* wg_b,
                         __bf16* qn, float* need, float* sal, float* gate) {
  __shared__ float r0[D_MOD], r1[D_MOD], r2[D_MOD], r3[D_MOD];
  int l = blockIdx.x, t = threadIdx.x;
  float xv = x[(size_t)l * D_MOD + t];
  r0[t] = xv * xv;
  r1[t] = xv * unc_w[t];
  r2[t] = xv * sal_w[t];
  r3[t] = xv * wg_w[t];
  __syncthreads();
  for (int s = D_MOD / 2; s > 0; s >>= 1) {
    if (t < s) { r0[t] += r0[t + s]; r1[t] += r1[t + s];
                 r2[t] += r2[t + s]; r3[t] += r3[t + s]; }
    __syncthreads();
  }
  float nrm = fmaxf(sqrtf(r0[0]), 1e-12f);
  qn[(size_t)l * D_MOD + t] = (__bf16)(xv / nrm);
  if (t == 0) {
    need[l] = 1.f / (1.f + __expf(-(r1[0] + unc_b[0])));
    sal[l]  = 1.f / (1.f + __expf(-(r2[0] + sal_b[0])));
    gate[l] = 1.f / (1.f + __expf(-(r3[0] + wg_b[0])));
  }
}

__global__ void k_knnorm(const float* mk, __bf16* kn) {
  __shared__ float red[D_MOD];
  int m = blockIdx.x, t = threadIdx.x;
  float v = mk[(size_t)m * D_MOD + t];
  red[t] = v * v;
  __syncthreads();
  for (int s = D_MOD / 2; s > 0; s >>= 1) {
    if (t < s) red[t] += red[t + s];
    __syncthreads();
  }
  float nrm = fmaxf(sqrtf(red[0]), 1e-12f);
  kn[(size_t)m * D_MOD + t] = (__bf16)(v / nrm);
}

// pass 1: per query-row running max & sum(exp) of 5*cos-sim over all M keys
__global__ void k_memstats(const __bf16* qn, const __bf16* kn,
                           float* sm, float* ss) {
  int wave = threadIdx.x >> 5;
  int l = threadIdx.x & 31;
  int g = l >> 4;
  int q0 = (blockIdx.x * 4 + wave) * 16;
  v16bf qa[8];
#pragma unroll
  for (int c = 0; c < 8; ++c) qa[c] = load_a_bf16(qn, D_MOD, q0, 32 * c);
  float mrun[8], srun[8];
#pragma unroll
  for (int i = 0; i < 8; ++i) { mrun[i] = -1e30f; srun[i] = 0.f; }
  for (int kb = 0; kb < M_MEM; kb += 32) {
    if (kb + 32 < M_MEM)  // prefetch next 32 key rows (one lane per row)
      __builtin_prefetch(kn + (size_t)(kb + 32 + l) * D_MOD, 0, 3);
    v8f s0 = {0, 0, 0, 0, 0, 0, 0, 0}, s1 = s0;
#pragma unroll
    for (int c = 0; c < 8; ++c) {
      s0 = wmma_bf16(qa[c], load_bT<__bf16>(kn, D_MOD, kb, 32 * c, M_MEM - 1), s0);
      s1 = wmma_bf16(qa[c], load_bT<__bf16>(kn, D_MOD, kb + 16, 32 * c, M_MEM - 1), s1);
    }
#pragma unroll
    for (int i = 0; i < 8; ++i) {
      float a = s0[i] * 5.f, b = s1[i] * 5.f;
      float mx = redmax16(fmaxf(a, b));
      float mnew = fmaxf(mrun[i], mx);
      float rs = redsum16(__expf(a - mnew) + __expf(b - mnew));
      srun[i] = srun[i] * __expf(mrun[i] - mnew) + rs;
      mrun[i] = mnew;
    }
  }
  if ((l & 15) == 0) {
#pragma unroll
    for (int i = 0; i < 8; ++i) {
      sm[q0 + i + 8 * g] = mrun[i];
      ss[q0 + i + 8 * g] = srun[i];
    }
  }
}

// pass 2: retrieved = softmax(5*res) @ mem_v, fused need-gate, per 64-col chunk
__global__ void k_memread(const __bf16* qn, const __bf16* kn, const float* mv,
                          const float* sm, const float* ss, const float* need,
                          __bf16* memin) {
  __shared__ float pbuf[4][16 * 32];
  int wave = threadIdx.x >> 5;
  int l = threadIdx.x & 31;
  int g = l >> 4;
  int col = l & 15;
  int q0 = (blockIdx.x * 4 + wave) * 16;
  int c0 = blockIdx.y * 64;
  float* pt = pbuf[wave];
  v16bf qa[8];
#pragma unroll
  for (int c = 0; c < 8; ++c) qa[c] = load_a_bf16(qn, D_MOD, q0, 32 * c);
  float mrow[8], isrow[8];
#pragma unroll
  for (int i = 0; i < 8; ++i) {
    int r = q0 + i + 8 * g;
    mrow[i] = sm[r];
    isrow[i] = 1.f / ss[r];
  }
  v8f o0 = {0, 0, 0, 0, 0, 0, 0, 0}, o1 = o0, o2 = o0, o3 = o0;
  for (int kb = 0; kb < M_MEM; kb += 32) {
    if (kb + 32 < M_MEM) {
      __builtin_prefetch(kn + (size_t)(kb + 32 + l) * D_MOD, 0, 3);
      __builtin_prefetch(mv + (size_t)(kb + 32 + l) * D_MOD + c0, 0, 3);
    }
    v8f s0 = {0, 0, 0, 0, 0, 0, 0, 0}, s1 = s0;
#pragma unroll
    for (int c = 0; c < 8; ++c) {
      s0 = wmma_bf16(qa[c], load_bT<__bf16>(kn, D_MOD, kb, 32 * c, M_MEM - 1), s0);
      s1 = wmma_bf16(qa[c], load_bT<__bf16>(kn, D_MOD, kb + 16, 32 * c, M_MEM - 1), s1);
    }
#pragma unroll
    for (int i = 0; i < 8; ++i) {
      pt[(i + 8 * g) * 32 + col]      = __expf(s0[i] * 5.f - mrow[i]) * isrow[i];
      pt[(i + 8 * g) * 32 + col + 16] = __expf(s1[i] * 5.f - mrow[i]) * isrow[i];
    }
    asm volatile("s_wait_dscnt 0" ::: "memory");
    v16bf pa = load_a_lds32(pt);
    o0 = wmma_bf16(pa, load_bKN<float>(mv, D_MOD, kb, c0), o0);
    o1 = wmma_bf16(pa, load_bKN<float>(mv, D_MOD, kb, c0 + 16), o1);
    o2 = wmma_bf16(pa, load_bKN<float>(mv, D_MOD, kb, c0 + 32), o2);
    o3 = wmma_bf16(pa, load_bKN<float>(mv, D_MOD, kb, c0 + 48), o3);
  }
#pragma unroll
  for (int i = 0; i < 8; ++i) {
    int row = q0 + i + 8 * g;
    float nv = need[row];
    size_t o = (size_t)row * D_MOD + c0 + col;
    memin[o]      = (__bf16)(nv * o0[i]);
    memin[o + 16] = (__bf16)(nv * o1[i]);
    memin[o + 32] = (__bf16)(nv * o2[i]);
    memin[o + 48] = (__bf16)(nv * o3[i]);
  }
}

// landmark gate-penalty (deterministic tree reduction), writes d_out[L*V]
__global__ void k_landmark(const float* sal, const float* gate, float* out) {
  __shared__ float sn[256], sc[256];
  int t = threadIdx.x;
  float num = 0.f, cnt = 0.f;
  for (int l = t; l < L_SEQ; l += 256) {
    float s = sal[l];
    float mx = -1e30f;
    for (int d = -15; d <= 15; ++d) {
      int j = l + d;
      if (j >= 0 && j < L_SEQ) mx = fmaxf(mx, sal[j]);
    }
    if (s == mx && s > 0.15f) {
      float gm = 1.f - gate[l];
      num += gm * gm;
      cnt += 1.f;
    }
  }
  sn[t] = num; sc[t] = cnt;
  __syncthreads();
  for (int s = 128; s > 0; s >>= 1) {
    if (t < s) { sn[t] += sn[t + s]; sc[t] += sc[t + s]; }
    __syncthreads();
  }
  if (t == 0)
    out[(size_t)L_SEQ * V_VOCAB] = (sc[0] > 0.f) ? sn[0] / fmaxf(sc[0], 1.f) : 0.f;
}

// ---------------------------------------------------------------------------
extern "C" void kernel_launch(void* const* d_in, const int* in_sizes, int n_in,
                              void* d_out, int out_size, void* d_ws, size_t ws_size,
                              hipStream_t stream) {
  const int*   tokens   = (const int*)d_in[0];
  const float* embed    = (const float*)d_in[1];
  const float* qkv_w    = (const float*)d_in[2];
  const float* qkv_b    = (const float*)d_in[3];
  const float* proj_w   = (const float*)d_in[4];
  const float* proj_b   = (const float*)d_in[5];
  const float* ln1_w    = (const float*)d_in[6];
  const float* mlp_w1   = (const float*)d_in[7];
  const float* mlp_b1   = (const float*)d_in[8];
  const float* mlp_w2   = (const float*)d_in[9];
  const float* mlp_b2   = (const float*)d_in[10];
  const float* ln2_w    = (const float*)d_in[11];
  const float* sal_w    = (const float*)d_in[12];
  const float* sal_b    = (const float*)d_in[13];
  const float* unc_w    = (const float*)d_in[14];
  const float* unc_b    = (const float*)d_in[15];
  const float* wg_w     = (const float*)d_in[16];
  const float* wg_b     = (const float*)d_in[17];
  const float* memproj_w = (const float*)d_in[18];
  const float* memproj_b = (const float*)d_in[19];
  const float* mem_k    = (const float*)d_in[20];
  const float* mem_v    = (const float*)d_in[21];
  float* logits = (float*)d_out;

  // workspace carve-up
  char* w = (char*)d_ws;
  size_t off = 0;
  auto alloc = [&](size_t bytes) {
    void* p = w + off;
    off = (off + bytes + 255) & ~(size_t)255;
    return p;
  };
  float*  x        = (float*)alloc((size_t)L_SEQ * D_MOD * 4);
  float*  x_out    = (float*)alloc((size_t)L_SEQ * D_MOD * 4);
  float*  qkv      = (float*)alloc((size_t)L_SEQ * 3 * D_MOD * 4);
  __bf16* xn       = (__bf16*)alloc((size_t)L_SEQ * D_MOD * 2);
  __bf16* mid_bf   = (__bf16*)alloc((size_t)L_SEQ * 4 * D_MOD * 2);
  __bf16* qh       = (__bf16*)alloc((size_t)L_SEQ * D_MOD * 2);
  __bf16* kh       = (__bf16*)alloc((size_t)L_SEQ * D_MOD * 2);
  __bf16* vh       = (__bf16*)alloc((size_t)L_SEQ * D_MOD * 2);
  __bf16* attn_bf  = (__bf16*)alloc((size_t)L_SEQ * D_MOD * 2);
  __bf16* qn       = (__bf16*)alloc((size_t)L_SEQ * D_MOD * 2);
  __bf16* knb      = (__bf16*)alloc((size_t)M_MEM * D_MOD * 2);
  __bf16* memin    = (__bf16*)alloc((size_t)L_SEQ * D_MOD * 2);
  __bf16* xout_bf  = (__bf16*)alloc((size_t)L_SEQ * D_MOD * 2);
  float*  need     = (float*)alloc((size_t)L_SEQ * 4);
  float*  sal      = (float*)alloc((size_t)L_SEQ * 4);
  float*  gate     = (float*)alloc((size_t)L_SEQ * 4);
  float*  statM    = (float*)alloc((size_t)L_SEQ * 4);
  float*  statS    = (float*)alloc((size_t)L_SEQ * 4);
  (void)ws_size; (void)in_sizes; (void)n_in; (void)out_size;

  auto gemm = [&](const __bf16* A, const float* W, const float* bias,
                  const float* res, float* oF, __bf16* oB,
                  int M, int N, int K, int gelu) {
    dim3 grid((N + 255) / 256, M / 16);
    k_gemm<<<grid, 256, 0, stream>>>(A, W, bias, res, oF, oB, M, N, K, gelu);
  };

  // embed gather
  k_gather<<<L_SEQ, D_MOD, 0, stream>>>(tokens, embed, x);

  for (int li = 0; li < N_LAYER; ++li) {
    const float* Wqkv = qkv_w + (size_t)li * 3 * D_MOD * D_MOD;
    const float* Bqkv = qkv_b + (size_t)li * 3 * D_MOD;
    const float* Wp   = proj_w + (size_t)li * D_MOD * D_MOD;
    const float* Bp   = proj_b + (size_t)li * D_MOD;
    const float* W1   = mlp_w1 + (size_t)li * 4 * D_MOD * D_MOD;
    const float* B1   = mlp_b1 + (size_t)li * 4 * D_MOD;
    const float* W2   = mlp_w2 + (size_t)li * D_MOD * 4 * D_MOD;
    const float* B2   = mlp_b2 + (size_t)li * D_MOD;

    k_rmsnorm<<<L_SEQ, D_MOD, 0, stream>>>(x, ln1_w + (size_t)li * D_MOD, xn);
    gemm(xn, Wqkv, Bqkv, nullptr, qkv, nullptr, L_SEQ, 3 * D_MOD, D_MOD, 0);
    k_rope_split<<<(L_SEQ * N_HEAD * 32 + 255) / 256, 256, 0, stream>>>(qkv, qh, kh, vh);
    k_attn<<<dim3(L_SEQ / 64, N_HEAD), 128, 0, stream>>>(qh, kh, vh, attn_bf);
    gemm(attn_bf, Wp, Bp, x, x, nullptr, L_SEQ, D_MOD, D_MOD, 0);  // in-place residual
    k_rmsnorm<<<L_SEQ, D_MOD, 0, stream>>>(x, ln2_w + (size_t)li * D_MOD, xn);
    gemm(xn, W1, B1, nullptr, nullptr, mid_bf, L_SEQ, 4 * D_MOD, D_MOD, 1);  // GELU
    gemm(mid_bf, W2, B2, x, x, nullptr, L_SEQ, D_MOD, 4 * D_MOD, 0);
  }

  // landmark memory read
  k_rowops<<<L_SEQ, D_MOD, 0, stream>>>(x, unc_w, unc_b, sal_w, sal_b, wg_w, wg_b,
                                        qn, need, sal, gate);
  k_knnorm<<<M_MEM, D_MOD, 0, stream>>>(mem_k, knb);
  k_memstats<<<L_SEQ / 64, 128, 0, stream>>>(qn, knb, statM, statS);
  k_memread<<<dim3(L_SEQ / 64, D_MOD / 64), 128, 0, stream>>>(
      qn, knb, mem_v, statM, statS, need, memin);
  gemm(memin, memproj_w, memproj_b, x, x_out, xout_bf, L_SEQ, D_MOD, D_MOD, 0);

  // tied-embedding logits (embed stays L2-resident: 51.5 MB < 192 MB L2)
  gemm(xout_bf, embed, nullptr, nullptr, logits, nullptr, L_SEQ, V_VOCAB, D_MOD, 0);

  // gate penalty scalar
  k_landmark<<<1, 256, 0, stream>>>(sal, gate, logits);
}
